// VariationalAutoencoder_17025250361623
// MI455X (gfx1250) — matrix operand order repaired and checked by reading, not compiled, not run on hardware
//
#include <hip/hip_runtime.h>

#define B_ 1024
#define M_ 4096
#define W_ 2
#define L_ 2
#define Z_ 64
#define H_ 8192

// GEMM workgroup tile
#define TM 128
#define TN 128
#define TK 32

typedef __attribute__((ext_vector_type(16))) _Float16 v16h;
typedef __attribute__((ext_vector_type(8)))  _Float16 v8h;
typedef __attribute__((ext_vector_type(8)))  float    v8f;
typedef __attribute__((ext_vector_type(4)))  float    v4f;

// Build a 16-element f16 fragment from two runs of 8 consecutive fp32 values.
__device__ __forceinline__ v16h pack16_from_f32(const float* p0, const float* p1) {
  v4f a0 = *(const v4f*)(p0);
  v4f a1 = *(const v4f*)(p0 + 4);
  v4f b0 = *(const v4f*)(p1);
  v4f b1 = *(const v4f*)(p1 + 4);
  v16h r;
#pragma unroll
  for (int i = 0; i < 4; ++i) {
    r[i]      = (_Float16)a0[i];
    r[4 + i]  = (_Float16)a1[i];
    r[8 + i]  = (_Float16)b0[i];
    r[12 + i] = (_Float16)b1[i];
  }
  return r;
}

// Same but from two runs of 8 consecutive f16 values.
__device__ __forceinline__ v16h pack16_from_f16(const _Float16* p0, const _Float16* p1) {
  v8h a = *(const v8h*)(p0);
  v8h b = *(const v8h*)(p1);
  v16h r;
#pragma unroll
  for (int i = 0; i < 8; ++i) { r[i] = a[i]; r[8 + i] = b[i]; }
  return r;
}

__device__ __forceinline__ float sigmoidf_(float x) {
  return 1.0f / (1.0f + __expf(-x));
}

// Staged load of 8 consecutive elements as f16 (converting if fp32 source).
template <bool F16>
__device__ __forceinline__ v8h stage_load(const void* __restrict__ src, size_t off) {
  if (F16) {
    return *(const v8h*)((const _Float16*)src + off);
  } else {
    const float* fp = (const float*)src + off;
    v4f f0 = *(const v4f*)(fp);
    v4f f1 = *(const v4f*)(fp + 4);
    v8h o;
#pragma unroll
    for (int j = 0; j < 4; ++j) { o[j] = (_Float16)f0[j]; o[4 + j] = (_Float16)f1[j]; }
    return o;
  }
}

// ---------------------------------------------------------------------------
// fp32 -> f16 stream conversion (8 elements/thread). n must be multiple of 8.
// ---------------------------------------------------------------------------
__global__ void __launch_bounds__(256)
vae_cvt_f16(const float* __restrict__ src, _Float16* __restrict__ dst, int n) {
  const int i = (blockIdx.x * 256 + threadIdx.x) * 8;
  if (i < n) {
    v4f a = *(const v4f*)(src + i);
    v4f b = *(const v4f*)(src + i + 4);
    v8h o;
#pragma unroll
    for (int j = 0; j < 4; ++j) { o[j] = (_Float16)a[j]; o[4 + j] = (_Float16)b[j]; }
    *(v8h*)(dst + i) = o;
  }
}

// ---------------------------------------------------------------------------
// h = relu(A @ W^T + bias), A:[rows,K], W:[O,K] (fp32 or f16 each),
// output f16 [rows,O]. LDS-staged, double-buffered:
// 256 threads (8 waves) per 128x128 tile. Per K-chunk of 32: prefetch next
// 128x32 A/B panels to regs, compute 8 WMMAs/wave from current LDS buffers,
// store prefetched regs into the other buffer, one barrier per chunk.
// grid = (O/TN, rows/TM), block = 256.
// ---------------------------------------------------------------------------
template <bool A_F16, bool B_F16>
__global__ void __launch_bounds__(256)
vae_gemm_relu(const void* __restrict__ Ap, const void* __restrict__ Wt,
              const float* __restrict__ bias, _Float16* __restrict__ Hout,
              int K, int Ocols) {
  __shared__ _Float16 Alds[2][TM * TK];   // [buf][row][k], row stride TK
  __shared__ _Float16 Blds[2][TN * TK];   // [buf][col][k], col stride TK

  const int tid  = threadIdx.x;        // 0..255
  const int lane = tid & 31;
  const int wave = tid >> 5;           // 0..7
  const int wr   = wave >> 1;          // 0..3 : 32-row group
  const int wc   = wave & 1;           // 0..1 : 64-col group

  const int mr  = lane & 15;
  const int hi  = lane >> 4;
  const int kbA = hi ? 16 : 0;         // A K-sub-base: contiguous k = 16*hi + s
  const int kbB = hi ? 16 : 0;         // B K-sub-base: contiguous k = 16*hi + s

  const int blockRow = blockIdx.y * TM;
  const int blockCol = blockIdx.x * TN;

  // staging: each thread copies 8 halves x 2 pieces for each of A and B
  const int sr0 = tid >> 2;            // row within panel (piece 0; piece 1 = +64)
  const int sc0 = (tid & 3) * 8;       // k-col within panel

  size_t rowA[2], rowB[2];
#pragma unroll
  for (int p = 0; p < 2; ++p) {
    rowA[p] = (size_t)(blockRow + sr0 + p * 64) * (size_t)K + sc0;
    rowB[p] = (size_t)(blockCol + sr0 + p * 64) * (size_t)K + sc0;
  }

  v8f acc[2][4] = {};
  v8h ar[2], br[2];

  // ---- prologue: stage chunk 0 into buffer 0 ----
#pragma unroll
  for (int p = 0; p < 2; ++p) {
    ar[p] = stage_load<A_F16>(Ap, rowA[p]);
    br[p] = stage_load<B_F16>(Wt, rowB[p]);
  }
#pragma unroll
  for (int p = 0; p < 2; ++p) {
    *(v8h*)&Alds[0][(sr0 + p * 64) * TK + sc0] = ar[p];
    *(v8h*)&Blds[0][(sr0 + p * 64) * TK + sc0] = br[p];
  }
  __syncthreads();

  const int nch = K / TK;
#pragma unroll 2
  for (int i = 0; i < nch; ++i) {
    const int cur = i & 1;
    const int nxt = cur ^ 1;
    const bool more = (i + 1) < nch;

    // prefetch next chunk into registers (overlaps with WMMA below)
    if (more) {
      const size_t koff = (size_t)(i + 1) * TK;
#pragma unroll
      for (int p = 0; p < 2; ++p) {
        ar[p] = stage_load<A_F16>(Ap, rowA[p] + koff);
        br[p] = stage_load<B_F16>(Wt, rowB[p] + koff);
      }
    }

    // compute from current LDS buffers: 8 WMMAs per wave
    {
      const _Float16* a0p = &Alds[cur][(wr * 32 + mr) * TK + kbA];
      const _Float16* a1p = a0p + 16 * TK;
      v16h a0 = pack16_from_f16(a0p, a0p + 8);
      v16h a1 = pack16_from_f16(a1p, a1p + 8);
#pragma unroll
      for (int t = 0; t < 4; ++t) {
        v16h b = *(const v16h*)&Blds[cur][(wc * 64 + t * 16 + mr) * TK + kbB];
        acc[0][t] = __builtin_amdgcn_wmma_f32_16x16x32_f16(
            false, a0, false, b, (short)0, acc[0][t], false, false);
        acc[1][t] = __builtin_amdgcn_wmma_f32_16x16x32_f16(
            false, a1, false, b, (short)0, acc[1][t], false, false);
      }
    }

    // commit prefetched chunk into the other buffer
    if (more) {
#pragma unroll
      for (int p = 0; p < 2; ++p) {
        *(v8h*)&Alds[nxt][(sr0 + p * 64) * TK + sc0] = ar[p];
        *(v8h*)&Blds[nxt][(sr0 + p * 64) * TK + sc0] = br[p];
      }
    }
    __syncthreads();
  }

  // ---- epilogue: bias + relu, f16 store ----
  const int row0 = blockRow + wr * 32;
  const int col0 = blockCol + wc * 64;
#pragma unroll
  for (int t = 0; t < 4; ++t) {
    const int c = col0 + t * 16 + mr;
    const float bvv = bias[c];
#pragma unroll
    for (int u = 0; u < 2; ++u) {
#pragma unroll
      for (int r = 0; r < 8; ++r) {
        const int bb = row0 + u * 16 + r + (hi ? 8 : 0);
        float v = acc[u][t][r] + bvv;
        v = v > 0.0f ? v : 0.0f;
        Hout[(size_t)bb * (size_t)Ocols + c] = (_Float16)v;
      }
    }
  }
}

// ---------------------------------------------------------------------------
// mu = h2 @ mu_w^T + mu_b ; lv = h2 @ lv_w^T + lv_b ; z = mu + exp(.5 lv)*eps
// 16x16 tile per wave, two accumulators share the A fragment.
// grid = (Z/16, B/16), block = 32.
// ---------------------------------------------------------------------------
__global__ void __launch_bounds__(32)
vae_head(const _Float16* __restrict__ h2, const float* __restrict__ mu_w,
         const float* __restrict__ mu_b, const float* __restrict__ lv_w,
         const float* __restrict__ lv_b, const float* __restrict__ eps,
         float* __restrict__ mu_out, float* __restrict__ lv_out,
         _Float16* __restrict__ z_out) {
  const int lane = threadIdx.x;
  const int row0 = blockIdx.y << 4;
  const int col0 = blockIdx.x << 4;
  const int mr   = lane & 15;
  const int hi   = lane >> 4;
  const int kbA  = hi ? 16 : 0;
  const int kbB  = hi ? 16 : 0;

  v8f cmu = {}, clv = {};
  const size_t arow = (size_t)(row0 + mr) * (size_t)M_;
  const size_t brow = (size_t)(col0 + mr) * (size_t)M_;

  for (int k0 = 0; k0 < M_; k0 += 32) {
    const _Float16* ap = h2 + arow + k0 + kbA;
    v16h a = pack16_from_f16(ap, ap + 8);
    const float* mp = mu_w + brow + k0 + kbB;
    v16h bm = pack16_from_f32(mp, mp + 8);
    cmu = __builtin_amdgcn_wmma_f32_16x16x32_f16(false, a, false, bm, (short)0, cmu, false, false);
    const float* lp = lv_w + brow + k0 + kbB;
    v16h bl = pack16_from_f32(lp, lp + 8);
    clv = __builtin_amdgcn_wmma_f32_16x16x32_f16(false, a, false, bl, (short)0, clv, false, false);
  }

  const int c = col0 + mr;
  const float mb = mu_b[c];
  const float lb = lv_b[c];
#pragma unroll
  for (int r = 0; r < 8; ++r) {
    const int bb = row0 + r + (hi ? 8 : 0);
    const size_t idx = (size_t)bb * Z_ + c;
    float mu = cmu[r] + mb;
    float lv = clv[r] + lb;
    mu_out[idx] = mu;
    lv_out[idx] = lv;
    float zz = mu + __expf(0.5f * lv) * eps[idx];
    z_out[idx] = (_Float16)zz;
  }
}

// ---------------------------------------------------------------------------
// One decoder branch, fully fused:
//   a = sigmoid(z @ ful_w^T + ful_b)            (WMMA over K=64)
//   for l in 0..1: a = sigmoid(blockdiag2x2(a)) (registers + shfl_xor)
//   out[b,m] = a0*ow0 + a1*ow1 + ob
// One wave handles a 16x16 tile of [B,H] = 16 batch rows x 8 measurements.
// grid = (H/16, B/16), block = 32. Launched twice (mean, cov).
// ---------------------------------------------------------------------------
__global__ void __launch_bounds__(32)
vae_decoder_branch(const _Float16* __restrict__ zin, const float* __restrict__ ful_w,
                   const float* __restrict__ ful_b, const float* __restrict__ hwp,
                   const float* __restrict__ hbp, const float* __restrict__ owp,
                   const float* __restrict__ obp, float* __restrict__ outp) {
  const int lane = threadIdx.x;
  const int row0 = blockIdx.y << 4;
  const int h0   = blockIdx.x << 4;
  const int mr   = lane & 15;
  const int hi   = lane >> 4;
  const int kbA  = hi ? 16 : 0;
  const int kbB  = hi ? 16 : 0;

  v8f acc = {};
#pragma unroll
  for (int k0 = 0; k0 < Z_; k0 += 32) {
    const _Float16* ap = zin + (size_t)(row0 + mr) * Z_ + k0 + kbA;
    v16h a = pack16_from_f16(ap, ap + 8);
    const float* wp = ful_w + (size_t)(h0 + mr) * Z_ + k0 + kbB;
    v16h b = pack16_from_f32(wp, wp + 8);
    acc = __builtin_amdgcn_wmma_f32_16x16x32_f16(false, a, false, b, (short)0, acc, false, false);
  }

  const int hcol = h0 + mr;             // hidden unit index for this lane
  const float fbias = ful_b[hcol];
  const int m = hcol >> 1;              // measurement index (lane pairs share m)
  // h_w[l,m,v,w] flat = (l*M + m)*4 + v*2 + w
  const v4f hw0 = *(const v4f*)(hwp + (size_t)m * 4);
  const v4f hw1 = *(const v4f*)(hwp + (size_t)(M_ + m) * 4);
  const float hb00 = hbp[2 * m], hb01 = hbp[2 * m + 1];
  const float hb10 = hbp[H_ + 2 * m], hb11 = hbp[H_ + 2 * m + 1];
  const float ow0 = owp[2 * m], ow1 = owp[2 * m + 1];
  const float obm = obp[m];
  const bool iseven = (mr & 1) == 0;

#pragma unroll
  for (int r = 0; r < 8; ++r) {
    float av = sigmoidf_(acc[r] + fbias);
    float pv = __shfl_xor(av, 1, 32);    // partner lane holds the other w
    float a0 = iseven ? av : pv;
    float a1 = iseven ? pv : av;
    float t0 = sigmoidf_(a0 * hw0[0] + a1 * hw0[1] + hb00);
    float t1 = sigmoidf_(a0 * hw0[2] + a1 * hw0[3] + hb01);
    float u0 = sigmoidf_(t0 * hw1[0] + t1 * hw1[1] + hb10);
    float u1 = sigmoidf_(t0 * hw1[2] + t1 * hw1[3] + hb11);
    float res = u0 * ow0 + u1 * ow1 + obm;
    if (iseven) {
      outp[(size_t)(row0 + r + (hi ? 8 : 0)) * M_ + m] = res;
    }
  }
}

// ---------------------------------------------------------------------------
extern "C" void kernel_launch(void* const* d_in, const int* in_sizes, int n_in,
                              void* d_out, int out_size, void* d_ws, size_t ws_size,
                              hipStream_t stream) {
  (void)in_sizes; (void)n_in; (void)out_size;

  const float* x       = (const float*)d_in[0];
  const float* eps     = (const float*)d_in[1];
  const float* enc1_w  = (const float*)d_in[2];
  const float* enc1_b  = (const float*)d_in[3];
  const float* enc2_w  = (const float*)d_in[4];
  const float* enc2_b  = (const float*)d_in[5];
  const float* mu_w    = (const float*)d_in[6];
  const float* mu_b    = (const float*)d_in[7];
  const float* lv_w    = (const float*)d_in[8];
  const float* lv_b    = (const float*)d_in[9];
  const float* m_ful_w = (const float*)d_in[10];
  const float* m_ful_b = (const float*)d_in[11];
  const float* m_h_w   = (const float*)d_in[12];
  const float* m_h_b   = (const float*)d_in[13];
  const float* m_out_w = (const float*)d_in[14];
  const float* m_out_b = (const float*)d_in[15];
  const float* c_ful_w = (const float*)d_in[16];
  const float* c_ful_b = (const float*)d_in[17];
  const float* c_h_w   = (const float*)d_in[18];
  const float* c_h_b   = (const float*)d_in[19];
  const float* c_out_w = (const float*)d_in[20];
  const float* c_out_b = (const float*)d_in[21];

  float* x_recon = (float*)d_out;
  float* logcov  = x_recon + (size_t)B_ * M_;
  float* mu_out  = logcov  + (size_t)B_ * M_;
  float* lv_out  = mu_out  + (size_t)B_ * Z_;

  // ws layout (prefix always used): h1 | h2 | z  [+ xf16 | w1f16 | w2f16]
  const size_t nBM = (size_t)B_ * M_;
  const size_t nMM = (size_t)M_ * M_;
  _Float16* h1  = (_Float16*)d_ws;
  _Float16* h2  = h1 + nBM;
  _Float16* zb  = h2 + nBM;
  _Float16* xf  = zb + (size_t)B_ * Z_;
  _Float16* w1f = xf + nBM;
  _Float16* w2f = w1f + nMM;
  const size_t full_bytes = (2 * nBM + (size_t)B_ * Z_ + nBM + 2 * nMM) * sizeof(_Float16);
  const bool fast = ws_size >= full_bytes;

  const dim3 blk32(32, 1, 1);
  const dim3 blk256(256, 1, 1);
  const dim3 gemm_grid(M_ / TN, B_ / TM);

  if (fast) {
    // One-time f16 conversion of x and the two 64MB encoder weights.
    vae_cvt_f16<<<dim3((B_ * M_ / 8 + 255) / 256), blk256, 0, stream>>>(x, xf, B_ * M_);
    vae_cvt_f16<<<dim3((M_ * M_ / 8 + 255) / 256), blk256, 0, stream>>>(enc1_w, w1f, M_ * M_);
    vae_cvt_f16<<<dim3((M_ * M_ / 8 + 255) / 256), blk256, 0, stream>>>(enc2_w, w2f, M_ * M_);
    vae_gemm_relu<true, true><<<gemm_grid, blk256, 0, stream>>>(
        (const void*)xf, (const void*)w1f, enc1_b, h1, M_, M_);
    vae_gemm_relu<true, true><<<gemm_grid, blk256, 0, stream>>>(
        (const void*)h1, (const void*)w2f, enc2_b, h2, M_, M_);
  } else {
    vae_gemm_relu<false, false><<<gemm_grid, blk256, 0, stream>>>(
        (const void*)x, (const void*)enc1_w, enc1_b, h1, M_, M_);
    vae_gemm_relu<true, false><<<gemm_grid, blk256, 0, stream>>>(
        (const void*)h1, (const void*)enc2_w, enc2_b, h2, M_, M_);
  }

  vae_head<<<dim3(Z_ / 16, B_ / 16), blk32, 0, stream>>>(
      h2, mu_w, mu_b, lv_w, lv_b, eps, mu_out, lv_out, zb);
  vae_decoder_branch<<<dim3(H_ / 16, B_ / 16), blk32, 0, stream>>>(
      zb, m_ful_w, m_ful_b, m_h_w, m_h_b, m_out_w, m_out_b, x_recon);
  vae_decoder_branch<<<dim3(H_ / 16, B_ / 16), blk32, 0, stream>>>(
      zb, c_ful_w, c_ful_b, c_h_w, c_h_b, c_out_w, c_out_b, logcov);
}
